// ActionFormerFPN_84095459656088
// MI455X (gfx1250) — matrix operand order repaired and checked by reading, not compiled, not run on hardware
//
#include <hip/hip_runtime.h>
#include <hip/hip_bf16.h>
#include <math.h>

// ---------------------------------------------------------------------------
// ActionFormer-style FPN on MI455X (gfx1250, wave32, WMMA)
// B=2, T=2304, C=512, H=8, hs=64, NB=4, strides {1,2,2,2}, window=19
// ---------------------------------------------------------------------------

typedef __attribute__((ext_vector_type(16))) __bf16 v16bf;
typedef __attribute__((ext_vector_type(8)))  __bf16 v8bf;
typedef __attribute__((ext_vector_type(4)))  __bf16 v4bf;
typedef __attribute__((ext_vector_type(8)))  float  v8f;

#define BATCH 2
#define CCH   512
#define TT0   2304
#define NHEAD 8
#define HS    64
#define CHID  2048
#define WOVR  9
#define NEGB  (-10000.0f)
#define ATT_SCALE 0.125f   /* 1/sqrt(64) */

// ===========================================================================
// 0. (B,T,D) -> (B,D,T) transpose through LDS tile
// ===========================================================================
__global__ __launch_bounds__(256) void xp_kernel(const float* __restrict__ src,
                                                 float* __restrict__ dst,
                                                 int T, int D) {
  __shared__ float tile[32][33];
  int b  = blockIdx.z;
  int t0 = blockIdx.x * 32;
  int d0 = blockIdx.y * 32;
  int tx = threadIdx.x, ty = threadIdx.y;
  #pragma unroll
  for (int i = 0; i < 4; ++i) {
    int t = t0 + ty + i * 8;
    tile[ty + i * 8][tx] = src[(size_t)b * T * D + (size_t)t * D + d0 + tx];
  }
  __syncthreads();
  #pragma unroll
  for (int i = 0; i < 4; ++i) {
    int d = d0 + ty + i * 8;
    dst[(size_t)b * D * T + (size_t)d * T + t0 + tx] = tile[tx][ty + i * 8];
  }
}

// ===========================================================================
// 1. valid masks per scale: valid[b,t] = !mask[b, t*fac]
// ===========================================================================
__global__ void valid_kernel(const unsigned char* __restrict__ mask,
                             int* __restrict__ v, int len, int fac, int T) {
  int i = blockIdx.x * blockDim.x + threadIdx.x;
  if (i >= BATCH * len) return;
  int b = i / len, t = i - b * len;
  v[i] = mask[b * T + t * fac] ? 0 : 1;
}

__global__ void maskout_kernel(const int* __restrict__ v, float* __restrict__ o, int n) {
  int i = blockIdx.x * blockDim.x + threadIdx.x;
  if (i < n) o[i] = v[i] ? 0.0f : 1.0f;
}

// ===========================================================================
// 2. channel LayerNorm over C=512 per (b,t) -- one wave per timestep,
//    wave32 butterfly reduction.
// ===========================================================================
__global__ __launch_bounds__(256) void ln_kernel(const float* __restrict__ x,
                                                 const float* __restrict__ w,
                                                 const float* __restrict__ bias,
                                                 float* __restrict__ y, int T) {
  int gid  = (blockIdx.x * blockDim.x + threadIdx.x) >> 5;
  int lane = threadIdx.x & 31;
  if (gid >= BATCH * T) return;
  int b = gid / T, t = gid - b * T;
  const float* xb = x + (size_t)b * CCH * T + t;
  float vals[16], s = 0.f, ss = 0.f;
  #pragma unroll
  for (int i = 0; i < 16; ++i) {
    float v = xb[(size_t)(lane + 32 * i) * T];
    vals[i] = v; s += v; ss += v * v;
  }
  #pragma unroll
  for (int m = 16; m; m >>= 1) { s += __shfl_xor(s, m, 32); ss += __shfl_xor(ss, m, 32); }
  float mu  = s * (1.0f / CCH);
  float var = ss * (1.0f / CCH) - mu * mu;
  float r   = rsqrtf(var + 1e-5f);
  float* yb = y + (size_t)b * CCH * T + t;
  #pragma unroll
  for (int i = 0; i < 16; ++i) {
    int c = lane + 32 * i;
    yb[(size_t)c * T] = (vals[i] - mu) * r * w[c] + bias[c];
  }
}

// ===========================================================================
// 3. fused depthwise conv3 (stride s, pad 1) + timestep mask + channel LN
// ===========================================================================
__global__ __launch_bounds__(256) void dwconv_ln_kernel(
    const float* __restrict__ x, const float* __restrict__ cw,
    const int* __restrict__ vnext, const float* __restrict__ nw,
    const float* __restrict__ nb, float* __restrict__ y,
    int Tin, int Tout, int stride) {
  int gid  = (blockIdx.x * blockDim.x + threadIdx.x) >> 5;
  int lane = threadIdx.x & 31;
  if (gid >= BATCH * Tout) return;
  int b = gid / Tout, t = gid - b * Tout;
  float mf = (float)vnext[b * Tout + t];
  int c0 = t * stride - 1;
  float vals[16], s = 0.f, ss = 0.f;
  #pragma unroll
  for (int i = 0; i < 16; ++i) {
    int c = lane + 32 * i;
    const float* xc = x + ((size_t)b * CCH + c) * Tin;
    float acc = 0.f;
    #pragma unroll
    for (int j = 0; j < 3; ++j) {
      int tt = c0 + j;
      if (tt >= 0 && tt < Tin) acc += cw[c * 3 + j] * xc[tt];
    }
    acc *= mf;
    vals[i] = acc; s += acc; ss += acc * acc;
  }
  #pragma unroll
  for (int m = 16; m; m >>= 1) { s += __shfl_xor(s, m, 32); ss += __shfl_xor(ss, m, 32); }
  float mu  = s * (1.0f / CCH);
  float var = ss * (1.0f / CCH) - mu * mu;
  float r   = rsqrtf(var + 1e-5f);
  float* yb = y + (size_t)b * CCH * Tout + t;
  #pragma unroll
  for (int i = 0; i < 16; ++i) {
    int c = lane + 32 * i;
    yb[(size_t)c * Tout] = (vals[i] - mu) * r * nw[c] + nb[c];
  }
}

// ===========================================================================
// 4. WMMA GEMM: Y[b,o,t] = sum_c W[o,c] * X[b,c,t] (+ fused epilogue)
//    128 threads = 4 waves; macrotile 64(M) x 32(N) x 64(K) per trip.
//    Each wave: one 16-row strip, two 16-col groups, 2 K-substeps
//    => 4 v_wmma_f32_16x16x32_bf16 per loop iteration.
//    LDS is laid out in fragment order:
//      sW[row][k]   (k contiguous)  -> A frag = 2x b128 LDS loads
//      sX[n][k]     (k contiguous)  -> B frag = 1x 32B LDS load
//    mode 0: +bias
//    mode 1: ( +bias + extra[pool] ) * mask      (proj + skip)
//    mode 2: gelu(+bias)                          (fc1)
//    mode 3: extra[res] + (+bias)*mask            (fc2 residual)
// ===========================================================================
__global__ __launch_bounds__(128) void gemm_wmma_kernel(
    const float* __restrict__ W, const float* __restrict__ X,
    const float* __restrict__ bias, float* __restrict__ Y,
    const float* __restrict__ extra, const int* __restrict__ msk,
    int Cout, int Cin, int T, int mode) {
  __shared__ __bf16 sW[64 * 64];   // 8 KB, [row][k]
  __shared__ __bf16 sX[32 * 64];   // 4 KB, [n][k]
  int b    = blockIdx.z;
  int m0   = blockIdx.y * 64;
  int t0   = blockIdx.x * 32;
  int tid  = threadIdx.x;
  int wv   = tid >> 5;
  int lane = tid & 31;
  int r    = lane & 15;
  int sel  = lane >> 4;
  const float* Xb = X + (size_t)b * Cin * T;
  v8f acc0 = {}, acc1 = {};
  for (int k0 = 0; k0 < Cin; k0 += 64) {
    // --- stage W tile (64x64 f32 -> bf16): 1024 float4 slots / 128 threads ---
    #pragma unroll
    for (int i = 0; i < 8; ++i) {
      int slot = tid * 8 + i;
      int rr = slot >> 4, kq = (slot & 15) << 2;
      const float4 w4 = *(const float4*)(W + (size_t)(m0 + rr) * Cin + k0 + kq);
      v4bf p;
      p[0] = (__bf16)w4.x; p[1] = (__bf16)w4.y; p[2] = (__bf16)w4.z; p[3] = (__bf16)w4.w;
      *(v4bf*)&sW[rr * 64 + kq] = p;   // 8B ds_store
    }
    // --- stage X tile (64K x 32T f32 -> bf16, transposed to [n][k]) ---
    #pragma unroll
    for (int i = 0; i < 4; ++i) {
      int slot = tid * 4 + i;
      int kk = slot >> 3, tq = (slot & 7) << 2;
      const float4 x4 = *(const float4*)(Xb + (size_t)(k0 + kk) * T + t0 + tq);
      sX[(tq + 0) * 64 + kk] = (__bf16)x4.x;
      sX[(tq + 1) * 64 + kk] = (__bf16)x4.y;
      sX[(tq + 2) * 64 + kk] = (__bf16)x4.z;
      sX[(tq + 3) * 64 + kk] = (__bf16)x4.w;
    }
    __syncthreads();
    // prefetch next K slab into cache while the matrix pipe works
    if (k0 + 64 < Cin) {
      __builtin_prefetch(W + (size_t)(m0 + (tid >> 1)) * Cin + (k0 + 64), 0, 1);
      __builtin_prefetch(Xb + (size_t)(k0 + 64 + lane) * T + t0, 0, 1);
      __builtin_prefetch(Xb + (size_t)(k0 + 96 + lane) * T + t0, 0, 1);
    }
    // --- 2 K-substeps x 2 column groups = 4 WMMAs ---
    const __bf16* wrow = &sW[(wv * 16 + r) * 64];
    #pragma unroll
    for (int ks = 0; ks < 2; ++ks) {
      // A frag: lanes 0-15 K 0..15, lanes 16-31 K 16..31 (matches B)
      v16bf a = *(const v16bf*)&wrow[ks * 32 + sel * 16];       // 32B ds load
      // B frag: lane half selects K 0..15 / 16..31; lane&15 = column
      v16bf b0 = *(const v16bf*)&sX[(r)      * 64 + ks * 32 + sel * 16]; // 32B ds load
      v16bf b1 = *(const v16bf*)&sX[(16 + r) * 64 + ks * 32 + sel * 16];
      acc0 = __builtin_amdgcn_wmma_f32_16x16x32_bf16(
          false, a, false, b0, (short)0, acc0, false, false);
      acc1 = __builtin_amdgcn_wmma_f32_16x16x32_bf16(
          false, a, false, b1, (short)0, acc1, false, false);
    }
    __syncthreads();
  }
  // --- fused epilogue; C/D layout: VGPR j -> row j + 8*(lane>=16), col lane&15
  int ta = t0 + r;
  int tb = t0 + 16 + r;
  float mfa = msk ? (float)msk[b * T + ta] : 1.0f;
  float mfb = msk ? (float)msk[b * T + tb] : 1.0f;
  #pragma unroll
  for (int j = 0; j < 8; ++j) {
    int m = m0 + wv * 16 + j + 8 * sel;
    float bm = bias[m];
    size_t ia = ((size_t)b * Cout + m) * T + ta;
    size_t ib = ((size_t)b * Cout + m) * T + tb;
    float va = acc0[j] + bm;
    float vb = acc1[j] + bm;
    if (mode == 1) {
      va = (va + extra[ia]) * mfa;
      vb = (vb + extra[ib]) * mfb;
    } else if (mode == 2) {
      va = 0.5f * va * (1.0f + erff(va * 0.70710678118f));
      vb = 0.5f * vb * (1.0f + erff(vb * 0.70710678118f));
    } else if (mode == 3) {
      va = extra[ia] + va * mfa;
      vb = extra[ib] + vb * mfb;
    }
    Y[ia] = va;
    Y[ib] = vb;
  }
}

// ===========================================================================
// 5. banded attention, window 19, one wave per (b,h,t); lanes split hs=64.
// ===========================================================================
__global__ __launch_bounds__(128) void attn_kernel(
    const float* __restrict__ q, const float* __restrict__ k,
    const float* __restrict__ v, const int* __restrict__ kvv,
    float* __restrict__ out, int T) {
  int warp = (blockIdx.x * blockDim.x + threadIdx.x) >> 5;
  int lane = threadIdx.x & 31;
  int total = BATCH * NHEAD * T;
  if (warp >= total) return;
  int b = warp / (NHEAD * T);
  int rem = warp - b * NHEAD * T;
  int h = rem / T, t = rem - h * T;
  size_t base = ((size_t)b * CCH + h * HS) * T;
  float q0 = q[base + (size_t)lane * T + t] * ATT_SCALE;
  float q1 = q[base + (size_t)(lane + 32) * T + t] * ATT_SCALE;
  float sc[19];
  #pragma unroll
  for (int w = 0; w < 19; ++w) {
    int ti = t + w - WOVR;
    int tc = ti < 0 ? 0 : (ti > T - 1 ? T - 1 : ti);
    float p = q0 * k[base + (size_t)lane * T + tc] +
              q1 * k[base + (size_t)(lane + 32) * T + tc];
    #pragma unroll
    for (int m = 16; m; m >>= 1) p += __shfl_xor(p, m, 32);
    bool inband = (ti >= 0) && (ti < T);
    float bv = kvv[b * T + tc] ? 0.0f : NEGB;
    sc[w] = inband ? (p + bv) : -INFINITY;
  }
  float mx = sc[0];
  #pragma unroll
  for (int w = 1; w < 19; ++w) mx = fmaxf(mx, sc[w]);
  float sum = 0.f;
  #pragma unroll
  for (int w = 0; w < 19; ++w) { float e = __expf(sc[w] - mx); sc[w] = e; sum += e; }
  float inv = (1.0f / sum) * (float)kvv[b * T + t];   // q-side mask
  float o0 = 0.f, o1 = 0.f;
  #pragma unroll
  for (int w = 0; w < 19; ++w) {
    int ti = t + w - WOVR;
    int tc = ti < 0 ? 0 : (ti > T - 1 ? T - 1 : ti);
    float a = sc[w] * inv;
    o0 += a * v[base + (size_t)lane * T + tc];
    o1 += a * v[base + (size_t)(lane + 32) * T + tc];
  }
  out[base + (size_t)lane * T + t] = o0;
  out[base + (size_t)(lane + 32) * T + t] = o1;
}

// ===========================================================================
// 6. max-pool window 3 stride 2 pad 1 (skip path for stride-2 blocks)
// ===========================================================================
__global__ void pool_kernel(const float* __restrict__ x, float* __restrict__ y,
                            int Tin, int Tout) {
  int i = blockIdx.x * blockDim.x + threadIdx.x;
  int total = BATCH * CCH * Tout;
  if (i >= total) return;
  int t = i % Tout, bc = i / Tout;
  const float* xp = x + (size_t)bc * Tin;
  float mx = -INFINITY;
  #pragma unroll
  for (int j = -1; j <= 1; ++j) {
    int tt = 2 * t + j;
    if (tt >= 0 && tt < Tin) mx = fmaxf(mx, xp[tt]);
  }
  y[i] = mx;
}

// ===========================================================================
// Host orchestration
// ===========================================================================
extern "C" void kernel_launch(void* const* d_in, const int* in_sizes, int n_in,
                              void* d_out, int out_size, void* d_ws, size_t ws_size,
                              hipStream_t stream) {
  const float*         src  = (const float*)d_in[0];
  const unsigned char* mask = (const unsigned char*)d_in[1];
  // d_in[2] = pos_embed: unused by the reference.
  auto P = [&](int blk, int j) -> const float* {
    // block dict insertion order:
    // 0 ln1_w 1 ln1_b 2 ln2_w 3 ln2_b 4 q_conv 5 k_conv 6 v_conv
    // 7 qn_w 8 qn_b 9 kn_w 10 kn_b 11 vn_w 12 vn_b
    // 13 q_w 14 q_b 15 k_w 16 k_b 17 v_w 18 v_b 19 proj_w 20 proj_b
    // 21 fc1_w 22 fc1_b 23 fc2_w 24 fc2_b
    return (const float*)d_in[3 + blk * 25 + j];
  };

  float* ws = (float*)d_ws;
  size_t off = 0;
  auto alloc = [&](size_t n) { float* p = ws + off; off += n; return p; };
  const size_t big = (size_t)BATCH * CCH * TT0;
  float* bx[3] = { alloc(big), alloc(big), alloc(big) };
  float* xln  = alloc(big);
  float* qc   = alloc(big);   // reused for attention output
  float* kc   = alloc(big);   // reused for pooled skip
  float* vc   = alloc(big);   // reused for ln2 output
  float* qb   = alloc(big);
  float* kb   = alloc(big);
  float* vb   = alloc(big);
  float* hbuf = alloc((size_t)BATCH * CHID * TT0);
  int* ivals = (int*)(ws + off);

  const int vlen[5] = { 2304, 2304, 1152, 576, 288 };
  const int vfac[5] = { 1, 1, 2, 4, 8 };
  int* varr[5];
  {
    size_t io = 0;
    for (int k = 0; k < 5; ++k) { varr[k] = ivals + io; io += (size_t)BATCH * vlen[k]; }
  }

  const int Tins[4]    = { 2304, 2304, 1152, 576 };
  const int Touts[4]   = { 2304, 1152, 576, 288 };
  const int strides[4] = { 1, 2, 2, 2 };

  float* d_outf = (float*)d_out;
  size_t feat_off = 0;
  size_t mask_base = 0;
  for (int i = 0; i < 4; ++i) mask_base += (size_t)BATCH * CCH * Touts[i];
  size_t mask_off = 0;

  // ---- input transpose (B,T,D)->(B,D,T) and valid masks per scale ----
  {
    dim3 g(TT0 / 32, CCH / 32, BATCH), blk(32, 8);
    xp_kernel<<<g, blk, 0, stream>>>(src, bx[0], TT0, CCH);
  }
  for (int k = 0; k < 5; ++k) {
    int n = BATCH * vlen[k];
    valid_kernel<<<(n + 255) / 256, 256, 0, stream>>>(mask, varr[k], vlen[k], vfac[k], TT0);
  }

  int ci = 0, oi = 1, ni = 2;
  for (int blk = 0; blk < 4; ++blk) {
    const int Tin = Tins[blk], Tout = Touts[blk], s = strides[blk];
    const int* vnxt = varr[blk + 1];
    float* cur = bx[ci];
    float* obf = bx[oi];
    float* nxf = bx[ni];

    // ln1
    {
      int warps = BATCH * Tin;
      ln_kernel<<<(warps + 7) / 8, 256, 0, stream>>>(cur, P(blk, 0), P(blk, 1), xln, Tin);
    }
    // dwconv3 + mask + LN for q/k/v
    {
      int warps = BATCH * Tout, g = (warps + 7) / 8;
      dwconv_ln_kernel<<<g, 256, 0, stream>>>(xln, P(blk, 4), vnxt, P(blk, 7),  P(blk, 8),  qc, Tin, Tout, s);
      dwconv_ln_kernel<<<g, 256, 0, stream>>>(xln, P(blk, 5), vnxt, P(blk, 9),  P(blk, 10), kc, Tin, Tout, s);
      dwconv_ln_kernel<<<g, 256, 0, stream>>>(xln, P(blk, 6), vnxt, P(blk, 11), P(blk, 12), vc, Tin, Tout, s);
    }
    // 1x1 conv GEMMs (512x512) -> q,k,v
    {
      dim3 g(Tout / 32, CCH / 64, BATCH);
      gemm_wmma_kernel<<<g, 128, 0, stream>>>(P(blk, 13), qc, P(blk, 14), qb, nullptr, nullptr, CCH, CCH, Tout, 0);
      gemm_wmma_kernel<<<g, 128, 0, stream>>>(P(blk, 15), kc, P(blk, 16), kb, nullptr, nullptr, CCH, CCH, Tout, 0);
      gemm_wmma_kernel<<<g, 128, 0, stream>>>(P(blk, 17), vc, P(blk, 18), vb, nullptr, nullptr, CCH, CCH, Tout, 0);
    }
    // banded attention -> qc (reuse)
    {
      int items = BATCH * NHEAD * Tout;
      attn_kernel<<<(items + 3) / 4, 128, 0, stream>>>(qb, kb, vb, vnxt, qc, Tout);
    }
    // skip path: pooled (s=2) or identity
    const float* poolp = cur;
    if (s > 1) {
      int n = BATCH * CCH * Tout;
      pool_kernel<<<(n + 255) / 256, 256, 0, stream>>>(cur, kc, Tin, Tout);
      poolp = kc;
    }
    // proj GEMM fused with (+pool)*mask -> obf
    {
      dim3 g(Tout / 32, CCH / 64, BATCH);
      gemm_wmma_kernel<<<g, 128, 0, stream>>>(P(blk, 19), qc, P(blk, 20), obf, poolp, vnxt, CCH, CCH, Tout, 1);
    }
    // ln2 -> vc (reuse)
    {
      int warps = BATCH * Tout;
      ln_kernel<<<(warps + 7) / 8, 256, 0, stream>>>(obf, P(blk, 2), P(blk, 3), vc, Tout);
    }
    // fc1 (2048x512) + GELU -> hbuf
    {
      dim3 g(Tout / 32, CHID / 64, BATCH);
      gemm_wmma_kernel<<<g, 128, 0, stream>>>(P(blk, 21), vc, P(blk, 22), hbuf, nullptr, nullptr, CHID, CCH, Tout, 2);
    }
    // fc2 (512x2048) + residual*mask -> nxf
    {
      dim3 g(Tout / 32, CCH / 64, BATCH);
      gemm_wmma_kernel<<<g, 128, 0, stream>>>(P(blk, 23), hbuf, P(blk, 24), nxf, obf, vnxt, CCH, CHID, Tout, 3);
    }
    // emit feature map + mask
    {
      size_t fsz = (size_t)BATCH * CCH * Tout;
      hipMemcpyAsync(d_outf + feat_off, nxf, fsz * sizeof(float),
                     hipMemcpyDeviceToDevice, stream);
      feat_off += fsz;
      int n = BATCH * Tout;
      maskout_kernel<<<(n + 255) / 256, 256, 0, stream>>>(vnxt, d_outf + mask_base + mask_off, n);
      mask_off += (size_t)n;
    }
    // rotate buffers: (cur,out,next) -> (next, cur, out)
    int t = ci; ci = ni; ni = oi; oi = t;
  }
  (void)n_in; (void)in_sizes; (void)out_size; (void)ws_size;
}
